// AxialSelfAttention2_80169859547956
// MI455X (gfx1250) — matrix operand order repaired and checked by reading, not compiled, not run on hardware
//
#include <hip/hip_runtime.h>

// ---------------------------------------------------------------------------
// AxialSelfAttention2 for MI455X (gfx1250, wave32, WMMA bf16 16x16x32)
// B=8, C=128, T=300, F=64, C4=32, CHUNK=16, CF=64, CT=sqrt(19200)
// ---------------------------------------------------------------------------

typedef __bf16 bf16x16 __attribute__((ext_vector_type(16)));
typedef float  f32x8   __attribute__((ext_vector_type(8)));

#define NB   8
#define NC   128
#define NT   300
#define NF   64
#define NC4  32
#define INV_CF 0.015625f                  // 1/64
#define INV_CT 7.216878364870322e-3f      // 1/sqrt(19200)
#define BN_EPS 1e-5f
#define NEG_BIG (-1.0e30f)

__device__ __forceinline__ f32x8 wmma_bf16(bf16x16 a, bf16x16 b, f32x8 c) {
  return __builtin_amdgcn_wmma_f32_16x16x32_bf16(false, a, false, b,
                                                 (short)0, c, false, false);
}

// Wave-local LDS ordering fence (store->load turnaround of per-wave scratch).
__device__ __forceinline__ void lds_wave_fence() {
  asm volatile("s_wait_dscnt 0" ::: "memory");
}

// ---- WMMA fragment builders (contiguous k order: element s of lane half h is k = 16h + s) ----------------
// A (16x32 bf16, MxK), source row-major [row][k], row stride ldm.
__device__ __forceinline__ bf16x16 load_frag_a(const __bf16* p, int row0,
                                               int ldm, int k0, int lane) {
  const int m = lane & 15, hi = lane >> 4;
  const __bf16* rp = p + (row0 + m) * ldm + k0;
  bf16x16 a;
#pragma unroll
  for (int v = 0; v < 8; ++v) {
    int kk = hi * 16 + 2 * v;
    a[2 * v]     = rp[kk];
    a[2 * v + 1] = rp[kk + 1];
  }
  return a;
}

// B (32x16 bf16, KxN), source stored transposed: [n][k], row stride ldm.
__device__ __forceinline__ bf16x16 load_frag_bT(const __bf16* p, int n0,
                                                int ldm, int k0, int lane) {
  const int n = lane & 15, hi = lane >> 4;
  const __bf16* rp = p + (n0 + n) * ldm + k0 + hi * 16;
  bf16x16 b;
#pragma unroll
  for (int v = 0; v < 8; ++v) {
    b[2 * v]     = rp[2 * v];
    b[2 * v + 1] = rp[2 * v + 1];
  }
  return b;
}

// B (32x16 bf16, KxN), source stored row-major [k][n], row stride ldm.
__device__ __forceinline__ bf16x16 load_frag_b(const __bf16* p, int k0,
                                               int ldm, int n0, int lane) {
  const int n = lane & 15, hi = lane >> 4;
  const __bf16* cp = p + (k0 + hi * 16) * ldm + n0 + n;
  bf16x16 b;
#pragma unroll
  for (int v = 0; v < 8; ++v) {
    b[2 * v]     = cp[(2 * v) * ldm];
    b[2 * v + 1] = cp[(2 * v + 1) * ldm];
  }
  return b;
}

// Reductions across the 16-lane half that owns one D-matrix row group.
__device__ __forceinline__ float half_max(float v) {
#pragma unroll
  for (int m = 1; m < 16; m <<= 1) v = fmaxf(v, __shfl_xor(v, m, 32));
  return v;
}
__device__ __forceinline__ float half_sum(float v) {
#pragma unroll
  for (int m = 1; m < 16; m <<= 1) v += __shfl_xor(v, m, 32);
  return v;
}

// ---------------------------------------------------------------------------
// Kernel 1: fused pointwise conv 1 & 2 (C->C4, BN, PReLU).
// Y[64 rows = W1||W2][n] = Wcat(64x128) @ X(128 x 64) per block; K=128.
// Writes qf (B,T,F,C4) and qt (B,F,T,C4) as bf16.
// grid = B*T (2400), block = 256 (8 waves).
// ---------------------------------------------------------------------------
__global__ __launch_bounds__(256) void k1_pointwise12(
    const float* __restrict__ x, const float* __restrict__ W1,
    const float* __restrict__ g1, const float* __restrict__ b1,
    const float* __restrict__ m1, const float* __restrict__ v1,
    const float* __restrict__ a1, const float* __restrict__ W2,
    const float* __restrict__ g2, const float* __restrict__ b2,
    const float* __restrict__ m2, const float* __restrict__ v2,
    const float* __restrict__ a2, __bf16* __restrict__ qf,
    __bf16* __restrict__ qt) {
  __shared__ alignas(16) __bf16 sW[64 * 128];   // [o][c]
  __shared__ alignas(16) __bf16 sXT[64 * 128];  // [n][c] (transposed X tile)
  __shared__ float sScale[64];
  __shared__ float sShift[64];

  const int tid = threadIdx.x;
  const int w = tid >> 5, lane = tid & 31;
  const int hi = lane >> 4, ln = lane & 15;
  const int b = blockIdx.x / NT;
  const int t = blockIdx.x % NT;
  const int n0 = t * NF;  // 64 positions = one full t row

  // Concatenated weights -> bf16 LDS.
  for (int i = tid; i < 64 * 128; i += 256) {
    int o = i >> 7, c = i & 127;
    float wv = (o < 32) ? W1[o * 128 + c] : W2[(o - 32) * 128 + c];
    sW[i] = (__bf16)wv;
  }
  // X tile (128 channels x 64 positions), store transposed as bf16.
  const float* xb = x + (size_t)b * NC * (NT * NF);
  for (int i = tid; i < 2048; i += 256) {
    int c = i >> 4, q = i & 15;
    float4 vv = *(const float4*)(xb + (size_t)c * (NT * NF) + n0 + q * 4);
    int nl = q * 4;
    sXT[(nl + 0) * 128 + c] = (__bf16)vv.x;
    sXT[(nl + 1) * 128 + c] = (__bf16)vv.y;
    sXT[(nl + 2) * 128 + c] = (__bf16)vv.z;
    sXT[(nl + 3) * 128 + c] = (__bf16)vv.w;
  }
  if (tid < 64) {
    float g, bb, mm, vv;
    if (tid < 32) { g = g1[tid]; bb = b1[tid]; mm = m1[tid]; vv = v1[tid]; }
    else { int o = tid - 32; g = g2[o]; bb = b2[o]; mm = m2[o]; vv = v2[o]; }
    float sc = g * rsqrtf(vv + BN_EPS);
    sScale[tid] = sc;
    sShift[tid] = bb - mm * sc;
  }
  __syncthreads();

  const float a1v = a1[0], a2v = a2[0];

#pragma unroll
  for (int tt = 0; tt < 2; ++tt) {
    int ti = w + 8 * tt;
    int mt = ti >> 2, nt = ti & 3;
    f32x8 acc{};
#pragma unroll
    for (int kc = 0; kc < 4; ++kc) {
      bf16x16 a  = load_frag_a(sW, 16 * mt, 128, 32 * kc, lane);
      bf16x16 bb = load_frag_bT(sXT, 16 * nt, 128, 32 * kc, lane);
      acc = wmma_bf16(a, bb, acc);
    }
    int fcol = 16 * nt + ln;
#pragma unroll
    for (int r = 0; r < 8; ++r) {
      int m = 16 * mt + 8 * hi + r;
      float y = acc[r] * sScale[m] + sShift[m];
      float al = (m < 32) ? a1v : a2v;
      y = (y >= 0.f) ? y : al * y;
      if (m < 32)
        qf[(((size_t)b * NT + t) * NF + fcol) * NC4 + m] = (__bf16)y;
      else
        qt[(((size_t)b * NF + fcol) * NT + t) * NC4 + (m - 32)] = (__bf16)y;
    }
  }
}

// ---------------------------------------------------------------------------
// Kernel 2: frequency attention (banded 64x64, per (b,t); one wave each).
// grid = B*T/4 (600), block = 128 (4 waves).
// ---------------------------------------------------------------------------
__global__ __launch_bounds__(128) void k2_freq_attn(
    const __bf16* __restrict__ qf, __bf16* __restrict__ outF) {
  __shared__ alignas(16) __bf16 sQ[4][64 * 32];  // [f][c]
  __shared__ alignas(16) __bf16 sP[4][64 * 64];  // softmax probs [f][g]

  const int tid = threadIdx.x;
  const int w = tid >> 5, lane = tid & 31;
  const int hi = lane >> 4, ln = lane & 15;
  const int bt = blockIdx.x * 4 + w;
  const int b = bt / NT, t = bt % NT;

  {  // wave-private copy of Q (contiguous 4KB block)
    const uint4* g = (const uint4*)(qf + (size_t)bt * (NF * NC4));
    uint4* s = (uint4*)sQ[w];
#pragma unroll
    for (int i = 0; i < 8; ++i) s[lane + 32 * i] = g[lane + 32 * i];
  }
  __syncthreads();

  // S = Q Q^T / CF, band mask |i-j|<=16, row softmax -> P (bf16 in LDS).
#pragma unroll
  for (int mt = 0; mt < 4; ++mt) {
    bf16x16 aq = load_frag_a(sQ[w], 16 * mt, 32, 0, lane);
    f32x8 s[4];
#pragma unroll
    for (int nt = 0; nt < 4; ++nt) {
      bf16x16 bk = load_frag_bT(sQ[w], 16 * nt, 32, 0, lane);
      f32x8 z{};
      s[nt] = wmma_bf16(aq, bk, z);
    }
#pragma unroll
    for (int r = 0; r < 8; ++r) {
      int M = 16 * mt + 8 * hi + r;
      float vv[4];
#pragma unroll
      for (int nt = 0; nt < 4; ++nt) {
        int n = 16 * nt + ln;
        int d = M - n; d = (d < 0) ? -d : d;
        vv[nt] = (d <= 16) ? s[nt][r] * INV_CF : NEG_BIG;
      }
      float mx = fmaxf(fmaxf(vv[0], vv[1]), fmaxf(vv[2], vv[3]));
      mx = half_max(mx);
      float p[4], sum = 0.f;
#pragma unroll
      for (int nt = 0; nt < 4; ++nt) { p[nt] = __expf(vv[nt] - mx); sum += p[nt]; }
      sum = half_sum(sum);
      float inv = 1.f / sum;
#pragma unroll
      for (int nt = 0; nt < 4; ++nt)
        sP[w][M * 64 + 16 * nt + ln] = (__bf16)(p[nt] * inv);
    }
  }
  lds_wave_fence();

  // O = P @ Q  (64x32); write out_f as (B,F,T,C4).
#pragma unroll
  for (int mt = 0; mt < 4; ++mt) {
    bf16x16 ap0 = load_frag_a(sP[w], 16 * mt, 64, 0, lane);
    bf16x16 ap1 = load_frag_a(sP[w], 16 * mt, 64, 32, lane);
#pragma unroll
    for (int nt = 0; nt < 2; ++nt) {
      f32x8 o{};
      o = wmma_bf16(ap0, load_frag_b(sQ[w], 0, 32, 16 * nt, lane), o);
      o = wmma_bf16(ap1, load_frag_b(sQ[w], 32, 32, 16 * nt, lane), o);
#pragma unroll
      for (int r = 0; r < 8; ++r) {
        int f = 16 * mt + 8 * hi + r;
        int c = 16 * nt + ln;
        outF[(((size_t)b * NF + f) * NT + t) * NC4 + c] = (__bf16)o[r];
      }
    }
  }
}

// ---------------------------------------------------------------------------
// Kernel 3: causal time attention, flash-style, per (b,f).
// grid = B*F (512), block = 256 (8 waves); wave handles query tiles w,w+8,w+16.
// ---------------------------------------------------------------------------
#define TPAD 320
__global__ __launch_bounds__(256) void k3_time_attn(
    const __bf16* __restrict__ qt, const __bf16* __restrict__ outF,
    __bf16* __restrict__ outT) {
  __shared__ alignas(16) __bf16 sQ[TPAD * 32];
  __shared__ alignas(16) __bf16 sV[TPAD * 32];
  __shared__ alignas(16) __bf16 sP[8][16 * 32];

  const int tid = threadIdx.x;
  const int w = tid >> 5, lane = tid & 31;
  const int hi = lane >> 4, ln = lane & 15;
  const int bf = blockIdx.x;  // b*64 + f

  {  // load Q and V (300x32 bf16 each, contiguous), zero-pad rows 300..319
    const unsigned int* gq = (const unsigned int*)(qt + (size_t)bf * (NT * NC4));
    const unsigned int* gv = (const unsigned int*)(outF + (size_t)bf * (NT * NC4));
    unsigned int* sq = (unsigned int*)sQ;
    unsigned int* sv = (unsigned int*)sV;
    for (int i = tid; i < 4800; i += 256) { sq[i] = gq[i]; sv[i] = gv[i]; }
    for (int i = tid; i < (TPAD - NT) * 16; i += 256) {
      sq[4800 + i] = 0u; sv[4800 + i] = 0u;
    }
  }
  __syncthreads();

  __bf16* sPw = sP[w];

  for (int qi = w; qi < 19; qi += 8) {
    const int i0 = qi * 16;
    bf16x16 aq = load_frag_a(sQ, i0, 32, 0, lane);
    f32x8 o0{}, o1{};
    float rmax[8], rsum[8];
#pragma unroll
    for (int r = 0; r < 8; ++r) { rmax[r] = -3.0e38f; rsum[r] = 0.f; }

    const int nch = (i0 + 16 + 31) >> 5;
    for (int jc = 0; jc < nch; ++jc) {
      const int jb = jc * 32;
      bf16x16 bk0 = load_frag_bT(sQ, jb, 32, 0, lane);
      bf16x16 bk1 = load_frag_bT(sQ, jb + 16, 32, 0, lane);
      f32x8 z{};
      f32x8 s0 = wmma_bf16(aq, bk0, z);
      f32x8 s1 = wmma_bf16(aq, bk1, z);
      const int j0 = jb + ln;
#pragma unroll
      for (int r = 0; r < 8; ++r) {
        int irow = i0 + 8 * hi + r;
        float v0 = (j0 <= irow) ? s0[r] * INV_CT : NEG_BIG;
        float v1 = (j0 + 16 <= irow) ? s1[r] * INV_CT : NEG_BIG;
        float mx = half_max(fmaxf(v0, v1));
        float newm = fmaxf(rmax[r], mx);
        float al = __expf(rmax[r] - newm);
        float p0 = __expf(v0 - newm);
        float p1 = __expf(v1 - newm);
        float cs = half_sum(p0 + p1);
        rsum[r] = rsum[r] * al + cs;
        rmax[r] = newm;
        o0[r] *= al;
        o1[r] *= al;
        sPw[(8 * hi + r) * 32 + ln]      = (__bf16)p0;
        sPw[(8 * hi + r) * 32 + ln + 16] = (__bf16)p1;
      }
      lds_wave_fence();
      bf16x16 ap  = load_frag_a(sPw, 0, 32, 0, lane);
      bf16x16 bv0 = load_frag_b(sV, jb, 32, 0, lane);
      bf16x16 bv1 = load_frag_b(sV, jb, 32, 16, lane);
      o0 = wmma_bf16(ap, bv0, o0);
      o1 = wmma_bf16(ap, bv1, o1);
    }

#pragma unroll
    for (int r = 0; r < 8; ++r) {
      int trow = i0 + 8 * hi + r;
      if (trow < NT) {
        float inv = 1.f / rsum[r];
        size_t base = ((size_t)bf * NT + trow) * NC4;
        outT[base + ln]      = (__bf16)(o0[r] * inv);
        outT[base + ln + 16] = (__bf16)(o1[r] * inv);
      }
    }
  }
}

// ---------------------------------------------------------------------------
// Kernel 4: pointwise conv 3 (C4->C), BN, PReLU, + residual (fp32 out).
// grid = B*T (2400), block = 256; wave w owns output-channel tile mt=w.
// ---------------------------------------------------------------------------
__global__ __launch_bounds__(256) void k4_pointwise3(
    const float* __restrict__ x, const __bf16* __restrict__ outT,
    const float* __restrict__ W3, const float* __restrict__ g3,
    const float* __restrict__ b3, const float* __restrict__ m3,
    const float* __restrict__ v3, const float* __restrict__ a3,
    float* __restrict__ out) {
  __shared__ alignas(16) __bf16 sW3[128 * 32];  // [c_out][k]
  __shared__ alignas(16) __bf16 sVt[64 * 32];   // [f][k]
  __shared__ float sScale[128];
  __shared__ float sShift[128];

  const int tid = threadIdx.x;
  const int w = tid >> 5, lane = tid & 31;
  const int hi = lane >> 4, ln = lane & 15;
  const int b = blockIdx.x / NT;
  const int t = blockIdx.x % NT;

  for (int i = tid; i < 128 * 32; i += 256) sW3[i] = (__bf16)W3[i];
  for (int i = tid; i < 64 * 32; i += 256) {
    int f = i >> 5, k = i & 31;
    sVt[i] = outT[(((size_t)b * NF + f) * NT + t) * NC4 + k];
  }
  if (tid < 128) {
    float sc = g3[tid] * rsqrtf(v3[tid] + BN_EPS);
    sScale[tid] = sc;
    sShift[tid] = b3[tid] - m3[tid] * sc;
  }
  __syncthreads();

  const float a3v = a3[0];
  const int mt = w;
  bf16x16 aw = load_frag_a(sW3, 16 * mt, 32, 0, lane);
#pragma unroll
  for (int nt = 0; nt < 4; ++nt) {
    f32x8 acc{};
    acc = wmma_bf16(aw, load_frag_bT(sVt, 16 * nt, 32, 0, lane), acc);
    int fcol = 16 * nt + ln;
#pragma unroll
    for (int r = 0; r < 8; ++r) {
      int c = 16 * mt + 8 * hi + r;
      float y = acc[r] * sScale[c] + sShift[c];
      y = (y >= 0.f) ? y : a3v * y;
      size_t idx = (((size_t)b * NC + c) * NT + t) * NF + fcol;
      out[idx] = y + x[idx];
    }
  }
}

// ---------------------------------------------------------------------------
extern "C" void kernel_launch(void* const* d_in, const int* in_sizes, int n_in,
                              void* d_out, int out_size, void* d_ws,
                              size_t ws_size, hipStream_t stream) {
  (void)in_sizes; (void)n_in; (void)out_size; (void)ws_size;
  const float* x  = (const float*)d_in[0];
  const float* W1 = (const float*)d_in[1];
  const float* g1 = (const float*)d_in[2];
  const float* b1 = (const float*)d_in[3];
  const float* m1 = (const float*)d_in[4];
  const float* v1 = (const float*)d_in[5];
  const float* a1 = (const float*)d_in[6];
  const float* W2 = (const float*)d_in[7];
  const float* g2 = (const float*)d_in[8];
  const float* b2 = (const float*)d_in[9];
  const float* m2 = (const float*)d_in[10];
  const float* v2 = (const float*)d_in[11];
  const float* a2 = (const float*)d_in[12];
  const float* W3 = (const float*)d_in[13];
  const float* g3 = (const float*)d_in[14];
  const float* b3 = (const float*)d_in[15];
  const float* m3 = (const float*)d_in[16];
  const float* v3 = (const float*)d_in[17];
  const float* a3 = (const float*)d_in[18];

  const size_t N1 = (size_t)NB * NT * NF * NC4;  // 4,915,200 elements
  __bf16* qf   = (__bf16*)d_ws;                  // (B,T,F,C4)
  __bf16* qt   = qf + N1;                        // (B,F,T,C4)
  __bf16* outF = qt + N1;                        // (B,F,T,C4)
  __bf16* outT = outF + N1;                      // (B,F,T,C4)

  k1_pointwise12<<<NB * NT, 256, 0, stream>>>(x, W1, g1, b1, m1, v1, a1, W2,
                                              g2, b2, m2, v2, a2, qf, qt);
  k2_freq_attn<<<(NB * NT) / 4, 128, 0, stream>>>(qf, outF);
  k3_time_attn<<<NB * NF, 256, 0, stream>>>(qt, outF, outT);
  k4_pointwise3<<<NB * NT, 256, 0, stream>>>(x, outT, W3, g3, b3, m3, v3, a3,
                                             (float*)d_out);
}
